// MinkEncoder_30081950941515
// MI455X (gfx1250) — matrix operand order repaired and checked by reading, not compiled, not run on hardware
//
#include <hip/hip_runtime.h>
#include <stdint.h>
#include <stddef.h>

typedef _Float16 h16;
typedef __attribute__((ext_vector_type(8)))  _Float16 v8h;
typedef __attribute__((ext_vector_type(16))) _Float16 v16h;
typedef __attribute__((ext_vector_type(8)))  float    v8f;

union V16U { v16h v; v8h h[2]; };

// ---------------------------------------------------------------- utilities
__global__ void k_zero(uint32_t* __restrict__ p, int nWords) {
  int i = blockIdx.x * blockDim.x + threadIdx.x;
  if (i < nWords) p[i] = 0u;
}

// ---------------------------------------------------------------- scatter points -> dense grid (channel-last, pad 4) + mask
__global__ void k_scatter(const int* __restrict__ cx, const int* __restrict__ cy,
                          const int* __restrict__ cz, const float* __restrict__ feats,
                          h16* __restrict__ grid, float* __restrict__ mask, int N) {
  int i = blockIdx.x * blockDim.x + threadIdx.x;
  if (i >= N) return;
  int v = (cx[i] * 256 + cy[i]) * 32 + cz[i];
  grid[(size_t)v * 4 + 0] = (h16)feats[i * 3 + 0];
  grid[(size_t)v * 4 + 1] = (h16)feats[i * 3 + 1];
  grid[(size_t)v * 4 + 2] = (h16)feats[i * 3 + 2];
  mask[v] = 1.0f;
}

// ---------------------------------------------------------------- stem conv 3->32, k=3 (VALU; Cin=3 too small for WMMA K)
__global__ void k_stem(const h16* __restrict__ grid, const float* __restrict__ w,
                       h16* __restrict__ out) {
  __shared__ float sw[32 * 3 * 27];
  for (int i = threadIdx.x; i < 32 * 3 * 27; i += blockDim.x) sw[i] = w[i];
  __syncthreads();
  int v = blockIdx.x * blockDim.x + threadIdx.x;   // V0 = 2^21, grid exact
  int z = v & 31, y = (v >> 5) & 255, x = v >> 13;
  float acc[32];
#pragma unroll
  for (int c = 0; c < 32; ++c) acc[c] = 0.f;
  for (int tap = 0; tap < 27; ++tap) {
    int xi = x + tap / 9 - 1, yi = y + (tap / 3) % 3 - 1, zi = z + tap % 3 - 1;
    if ((unsigned)xi < 256u && (unsigned)yi < 256u && (unsigned)zi < 32u) {
      size_t vi = ((size_t)((xi << 8) + yi) << 5) + zi;
      float f0 = (float)grid[vi * 4 + 0];
      float f1 = (float)grid[vi * 4 + 1];
      float f2 = (float)grid[vi * 4 + 2];
#pragma unroll
      for (int c = 0; c < 32; ++c) {
        const float* wc = &sw[c * 81 + tap];   // (O,I,3,3,3): o*81 + i*27 + tap
        acc[c] = fmaf(wc[0],  f0, acc[c]);
        acc[c] = fmaf(wc[27], f1, acc[c]);
        acc[c] = fmaf(wc[54], f2, acc[c]);
      }
    }
  }
  h16* o = out + (size_t)v * 32;
#pragma unroll
  for (int c = 0; c < 32; ++c) o[c] = (h16)acc[c];
}

// ---------------------------------------------------------------- repack (O,I,k,k,k) f32 -> B-fragment-ordered f16 tiles
// dst[((nTile*nChunks + chunk)*32 + lane)*16 + j]; lanes 0-15 carry K 0..15, 16-31 carry K 16..31 (contiguous k order, same as the A fragments)
__global__ void k_repack(const float* __restrict__ w, h16* __restrict__ dst,
                         int Cin, int taps, int total) {
  int idx = blockIdx.x * blockDim.x + threadIdx.x;
  if (idx >= total) return;
  int j = idx & 15;
  int lane = (idx >> 4) & 31;
  int rest = idx >> 9;                      // nTile*nChunks + chunk
  int nChunks = taps * (Cin >> 5);
  int chunk = rest % nChunks;
  int tile  = rest / nChunks;
  int n = tile * 16 + (lane & 15);
  int g = lane >> 4;
  int K = chunk * 32 + g * 16 + j;
  int tap = K / Cin, cin = K % Cin;
  dst[idx] = (h16)w[((size_t)n * Cin + cin) * taps + tap];
}

// ---------------------------------------------------------------- generic implicit-GEMM conv via v_wmma_f32_16x16x32_f16
// One wave computes a 16-voxel x 32-channel tile (2 accumulators share each A fragment).
// blockIdx.x * 8 waves -> M tiles; blockIdx.y -> pair of 16-channel N tiles.
// K (kernel size) and CPT (Cin/32 chunks per tap) are compile-time: inner loops unroll,
// b128 loads batch into clauses ahead of the WMMA chain.
template <int K, int CPT>
__global__ __launch_bounds__(256)
void k_conv_wmma(const h16* __restrict__ in, const h16* __restrict__ wpk,
                 h16* __restrict__ out,
                 int X, int Y, int Z,
                 int lYo, int lZo, int mTiles,
                 int Cout, int koff,
                 int sx, int sy, int sz) {
  constexpr int TAPS = K * K * K;
  constexpr int CIN  = CPT * 32;
  const int lane = threadIdx.x & 31;
  int mt = blockIdx.x * (blockDim.x >> 5) + (threadIdx.x >> 5);
  if (mt >= mTiles) return;                      // uniform per wave -> EXEC all-ones at WMMA
  const int nt = blockIdx.y << 1;
  const int g  = lane >> 4;
  int v  = (mt << 4) + (lane & 15);
  int zo = v & ((1 << lZo) - 1);
  int yo = (v >> lZo) & ((1 << lYo) - 1);
  int xo = v >> (lZo + lYo);
  const size_t wstride = (size_t)TAPS * CPT * 512;
  const h16* w0 = wpk + (size_t)nt * wstride + (size_t)lane * 16;
  const h16* w1 = w0 + wstride;
  v8f acc0 = {}, acc1 = {};
  v8h z8 = {};
#pragma unroll(K <= 2 ? TAPS : 1)
  for (int tap = 0; tap < TAPS; ++tap) {
    int xi = xo * sx + (tap / (K * K)) + koff;
    int yi = yo * sy + ((tap / K) % K) + koff;
    int zi = zo * sz + (tap % K) + koff;
    bool ok = ((unsigned)xi < (unsigned)X) && ((unsigned)yi < (unsigned)Y) &&
              ((unsigned)zi < (unsigned)Z);
    const h16* ap = in + (size_t)((xi * Y + yi) * Z + zi) * CIN + g * 16;
    const h16* bp0 = w0 + (size_t)tap * CPT * 512;
    const h16* bp1 = w1 + (size_t)tap * CPT * 512;
    V16U a[CPT], b0[CPT], b1[CPT];
#pragma unroll
    for (int cc = 0; cc < CPT; ++cc) {           // batched loads
      if (ok) {                                  // A layout: lane half g holds K=g*16..g*16+15 contiguous
        a[cc].h[0] = *(const v8h*)(ap + cc * 32);
        a[cc].h[1] = *(const v8h*)(ap + cc * 32 + 8);
      } else { a[cc].h[0] = z8; a[cc].h[1] = z8; }
      b0[cc].h[0] = *(const v8h*)(bp0 + cc * 512);
      b0[cc].h[1] = *(const v8h*)(bp0 + cc * 512 + 8);
      b1[cc].h[0] = *(const v8h*)(bp1 + cc * 512);
      b1[cc].h[1] = *(const v8h*)(bp1 + cc * 512 + 8);
    }
#pragma unroll
    for (int cc = 0; cc < CPT; ++cc) {           // WMMA chain
      acc0 = __builtin_amdgcn_wmma_f32_16x16x32_f16(
          false, a[cc].v, false, b0[cc].v, (short)0, acc0, false, false);
      acc1 = __builtin_amdgcn_wmma_f32_16x16x32_f16(
          false, a[cc].v, false, b1[cc].v, (short)0, acc1, false, false);
    }
  }
  // D layout: VGPR j -> M = j + g*8, N = lane%16
  h16* op = out + (size_t)(mt << 4) * Cout + (nt << 4) + (lane & 15);
#pragma unroll
  for (int j = 0; j < 8; ++j) {
    op[(size_t)(g * 8 + j) * Cout]      = (h16)acc0[j];
    op[(size_t)(g * 8 + j) * Cout + 16] = (h16)acc1[j];
  }
}

// ---------------------------------------------------------------- masked BN statistics: stats = {sum[C], sumsq[C], cnt}
__global__ void k_stats(const h16* __restrict__ x, const float* __restrict__ mask,
                        float* __restrict__ stats, int V, int lC) {
  __shared__ float sSum[128], sSq[128], sCnt;
  const int C = 1 << lC;
  int t = threadIdx.x;
  if (t < C) { sSum[t] = 0.f; sSq[t] = 0.f; }
  if (t == 0) sCnt = 0.f;
  __syncthreads();
  int c = t & (C - 1);
  int r = t >> lC;
  int R = (int)blockDim.x >> lC;
  float fs = 0.f, fq = 0.f, fc = 0.f;
  for (int v = blockIdx.x * R + r; v < V; v += gridDim.x * R) {
    float m = mask[v];
    float val = (float)x[((size_t)v << lC) + c] * m;
    fs += val; fq += val * val;
    if (c == 0) fc += m;
  }
  atomicAdd(&sSum[c], fs);
  atomicAdd(&sSq[c], fq);
  if (c == 0) atomicAdd(&sCnt, fc);
  __syncthreads();
  if (t < C) { atomicAdd(&stats[t], sSum[t]); atomicAdd(&stats[C + t], sSq[t]); }
  if (t == 0) atomicAdd(&stats[2 * C], sCnt);
}

__global__ void k_finalize(const float* __restrict__ stats, const float* __restrict__ g,
                           const float* __restrict__ b, float* __restrict__ sb, int C) {
  int c = threadIdx.x;
  if (c >= C) return;
  float cnt  = fmaxf(stats[2 * C], 1.0f);
  float mean = stats[c] / cnt;
  float var  = fmaxf(stats[C + c] / cnt - mean * mean, 0.f);
  float s = g[c] * rsqrtf(var + 1e-5f);
  sb[c] = s;
  sb[C + c] = b[c] - mean * s;
}

// y = [relu](s*x+b) * mask
__global__ void k_bn(const h16* __restrict__ x, const float* __restrict__ sb,
                     const float* __restrict__ mask, h16* __restrict__ out,
                     long long E, int lC, int relu) {
  long long e = (long long)blockIdx.x * blockDim.x + threadIdx.x;
  if (e >= E) return;
  int C = 1 << lC;
  int c = (int)e & (C - 1); long long v = e >> lC;
  float y = sb[c] * (float)x[e] + sb[C + c];
  if (relu) y = fmaxf(y, 0.f);
  out[e] = (h16)(y * mask[v]);
}

// y = relu((s*x+b)*m + sk) * m  (sk already masked, m in {0,1})
__global__ void k_bn_add_relu(const h16* __restrict__ x, const h16* __restrict__ sk,
                              const float* __restrict__ sb, const float* __restrict__ mask,
                              h16* __restrict__ out, long long E, int lC) {
  long long e = (long long)blockIdx.x * blockDim.x + threadIdx.x;
  if (e >= E) return;
  int C = 1 << lC;
  int c = (int)e & (C - 1); long long v = e >> lC;
  float y = sb[c] * (float)x[e] + sb[C + c] + (float)sk[e];
  out[e] = (h16)(fmaxf(y, 0.f) * mask[v]);
}

// occupancy max-pool, window == stride
__global__ void k_maskpool(const float* __restrict__ mi, float* __restrict__ mo,
                           int Y, int Z, int lYo, int lZo, int Vo,
                           int sx, int sy, int sz) {
  int v = blockIdx.x * blockDim.x + threadIdx.x;
  if (v >= Vo) return;
  int zo = v & ((1 << lZo) - 1);
  int yo = (v >> lZo) & ((1 << lYo) - 1);
  int xo = v >> (lZo + lYo);
  float m = 0.f;
  for (int a = 0; a < sx; ++a)
    for (int b = 0; b < sy; ++b)
      for (int c = 0; c < sz; ++c)
        m = fmaxf(m, mi[((xo * sx + a) * Y + yo * sy + b) * Z + zo * sz + c]);
  mo[v] = m;
}

// f16 channel-last -> f32 channel-major output [C][V]
__global__ void k_final(const h16* __restrict__ x, float* __restrict__ out,
                        int lV, int total, int lC) {
  int e = blockIdx.x * blockDim.x + threadIdx.x;
  if (e >= total) return;
  int c = e >> lV, v = e & ((1 << lV) - 1);
  out[e] = (float)x[((size_t)v << lC) + c];
}

// ================================================================ host
static inline int ilog2(int x) { return 31 - __builtin_clz((unsigned)x); }

extern "C" void kernel_launch(void* const* d_in, const int* in_sizes, int n_in,
                              void* d_out, int out_size, void* d_ws, size_t ws_size,
                              hipStream_t stream) {
  (void)n_in; (void)out_size; (void)ws_size;
  const int* cx = (const int*)d_in[0];
  const int* cy = (const int*)d_in[1];
  const int* cz = (const int*)d_in[2];
  const float* feats = (const float*)d_in[4];
  const int N = in_sizes[0];
  // params flattened as JAX pytree (sorted dict keys), leaves start at d_in[5]
  auto P = [&](int i) -> const float* { return (const float*)d_in[5 + i]; };

  const int V0 = 256 * 256 * 32;   // 2,097,152
  const int V1 = 128 * 128 * 32;   //   524,288
  const int V2 = 64 * 64 * 16;     //    65,536
  const int V3 = 32 * 32 * 8;      //     8,192

  // ---- workspace arena
  char* basep = (char*)d_ws;
  size_t aoff = 0;
  auto alloc = [&](size_t bytes) -> void* {
    void* p = basep + aoff;
    aoff += (bytes + 255) & ~(size_t)255;
    return p;
  };
  h16*   grid  = (h16*)  alloc((size_t)V0 * 4 * 2);
  float* mask0 = (float*)alloc((size_t)V0 * 4);
  float* mask1 = (float*)alloc((size_t)V1 * 4);
  float* mask2 = (float*)alloc((size_t)V2 * 4);
  float* mask3 = (float*)alloc((size_t)V3 * 4);
  h16* big0 = (h16*)alloc((size_t)V0 * 32 * 2);   // 128MB
  h16* big1 = (h16*)alloc((size_t)V0 * 32 * 2);   // 128MB
  h16* p0   = (h16*)alloc((size_t)V1 * 64 * 2);   // 64MB
  h16* p1   = (h16*)alloc((size_t)V1 * 64 * 2);
  h16* p2   = (h16*)alloc((size_t)V1 * 64 * 2);
  h16* wpkAll = (h16*)alloc(8u * 1024 * 1024);
  float* stats = (float*)alloc(4096);
  float* sb    = (float*)alloc(1024);
  h16* b0a = big0; h16* b0b = big0 + (size_t)V1 * 64;
  h16* b1a = big1; h16* b1b = big1 + (size_t)V1 * 64;

  auto zero = [&](void* p, size_t bytes) {
    int words = (int)(bytes >> 2);
    k_zero<<<(words + 255) / 256, 256, 0, stream>>>((uint32_t*)p, words);
  };
  size_t woff = 0;
  auto repack = [&](int leaf, int Cout, int Cin, int k) -> h16* {
    int taps = k * k * k;
    int total = Cout * taps * Cin;
    h16* dst = wpkAll + woff; woff += (size_t)total;
    k_repack<<<(total + 255) / 256, 256, 0, stream>>>(P(leaf), dst, Cin, taps, total);
    return dst;
  };
  auto conv = [&](const h16* in, const h16* wp, h16* outp,
                  int X, int Y, int Z, int Xo, int Yo, int Zo,
                  int Cin, int Cout, int k, int koff, int sx, int sy, int sz) {
    int mTiles = (Xo * Yo * Zo) >> 4;
    dim3 grd((mTiles + 7) / 8, Cout >> 5);           // 2 N-tiles per wave
    int lYo = ilog2(Yo), lZo = ilog2(Zo);
    int cpt = Cin >> 5;
    switch (k * 10 + cpt) {
      case 11: k_conv_wmma<1,1><<<grd,256,0,stream>>>(in,wp,outp,X,Y,Z,lYo,lZo,mTiles,Cout,koff,sx,sy,sz); break;
      case 12: k_conv_wmma<1,2><<<grd,256,0,stream>>>(in,wp,outp,X,Y,Z,lYo,lZo,mTiles,Cout,koff,sx,sy,sz); break;
      case 14: k_conv_wmma<1,4><<<grd,256,0,stream>>>(in,wp,outp,X,Y,Z,lYo,lZo,mTiles,Cout,koff,sx,sy,sz); break;
      case 21: k_conv_wmma<2,1><<<grd,256,0,stream>>>(in,wp,outp,X,Y,Z,lYo,lZo,mTiles,Cout,koff,sx,sy,sz); break;
      case 22: k_conv_wmma<2,2><<<grd,256,0,stream>>>(in,wp,outp,X,Y,Z,lYo,lZo,mTiles,Cout,koff,sx,sy,sz); break;
      case 24: k_conv_wmma<2,4><<<grd,256,0,stream>>>(in,wp,outp,X,Y,Z,lYo,lZo,mTiles,Cout,koff,sx,sy,sz); break;
      case 31: k_conv_wmma<3,1><<<grd,256,0,stream>>>(in,wp,outp,X,Y,Z,lYo,lZo,mTiles,Cout,koff,sx,sy,sz); break;
      case 32: k_conv_wmma<3,2><<<grd,256,0,stream>>>(in,wp,outp,X,Y,Z,lYo,lZo,mTiles,Cout,koff,sx,sy,sz); break;
      case 34: k_conv_wmma<3,4><<<grd,256,0,stream>>>(in,wp,outp,X,Y,Z,lYo,lZo,mTiles,Cout,koff,sx,sy,sz); break;
      default: break;
    }
  };
  auto bn = [&](const h16* x, int V, int C, int leaf_b, int leaf_g,
                const float* m, h16* outp, bool relu) {
    zero(stats, (size_t)(2 * C + 1) * 4);
    k_stats<<<256, 256, 0, stream>>>(x, m, stats, V, ilog2(C));
    k_finalize<<<1, 128, 0, stream>>>(stats, P(leaf_g), P(leaf_b), sb, C);
    long long E = (long long)V * C;
    k_bn<<<(int)((E + 255) / 256), 256, 0, stream>>>(x, sb, m, outp, E, ilog2(C), relu ? 1 : 0);
  };
  auto bnAdd = [&](const h16* x, const h16* sk, int V, int C, int leaf_b, int leaf_g,
                   const float* m, h16* outp) {
    zero(stats, (size_t)(2 * C + 1) * 4);
    k_stats<<<256, 256, 0, stream>>>(x, m, stats, V, ilog2(C));
    k_finalize<<<1, 128, 0, stream>>>(stats, P(leaf_g), P(leaf_b), sb, C);
    long long E = (long long)V * C;
    k_bn_add_relu<<<(int)((E + 255) / 256), 256, 0, stream>>>(x, sk, sb, m, outp, E, ilog2(C));
  };

  // ---- repack all weights (leaf indices per sorted pytree: s1,s2,s3,stem)
  h16* w_d1   = repack(2, 32, 32, 2);
  h16* w_r11  = repack(9, 64, 32, 3);
  h16* w_r12  = repack(10, 64, 64, 3);
  h16* w_r1d  = repack(11, 64, 32, 1);
  h16* w_r21  = repack(16, 64, 64, 3);
  h16* w_r22  = repack(17, 64, 64, 3);
  h16* w_d2   = repack(20, 64, 64, 2);
  h16* w2_r11 = repack(27, 128, 64, 3);
  h16* w2_r12 = repack(28, 128, 128, 3);
  h16* w2_r1d = repack(29, 128, 64, 1);
  h16* w2_r21 = repack(34, 128, 128, 3);
  h16* w2_r22 = repack(35, 128, 128, 3);
  h16* w_d3   = repack(38, 128, 128, 2);
  h16* w3_r11 = repack(43, 128, 128, 3);
  h16* w3_r12 = repack(44, 128, 128, 3);
  h16* w3_r21 = repack(49, 128, 128, 3);
  h16* w3_r22 = repack(50, 128, 128, 3);

  // ---- stem
  zero(grid, (size_t)V0 * 4 * 2);
  zero(mask0, (size_t)V0 * 4);
  k_scatter<<<(N + 255) / 256, 256, 0, stream>>>(cx, cy, cz, feats, grid, mask0, N);
  k_stem<<<V0 / 256, 256, 0, stream>>>(grid, P(53), big0);
  bn(big0, V0, 32, 51, 52, mask0, big1, true);          // A0 in big1

  // ---- one stage: down(k=2) + res1(+wd) + res2 ; output act lands in p1
  auto stage = [&](const h16* inAct, const float* mi, float* mo,
                   int Xi, int Yi, int Zi, int Xo, int Yo, int Zo,
                   int sx, int sy, int sz, int Cin, int Cr,
                   int d_b, int d_g, h16* d_w,
                   int r1b1, int r1g1, int r1b2, int r1g2, int r1bd, int r1gd,
                   h16* r1w1, h16* r1w2, h16* r1wd,
                   int r2b1, int r2g1, int r2b2, int r2g2, h16* r2w1, h16* r2w2) {
    int V = Xo * Yo * Zo;
    k_maskpool<<<(V + 255) / 256, 256, 0, stream>>>(mi, mo, Yi, Zi,
                                                    ilog2(Yo), ilog2(Zo), V, sx, sy, sz);
    conv(inAct, d_w, p0, Xi, Yi, Zi, Xo, Yo, Zo, Cin, Cin, 2, 0, sx, sy, sz);
    bn(p0, V, Cin, d_b, d_g, mo, p2, true);             // D in p2
    // res1: Cin -> Cr
    conv(p2, r1w1, b0a, Xo, Yo, Zo, Xo, Yo, Zo, Cin, Cr, 3, -1, 1, 1, 1);
    bn(b0a, V, Cr, r1b1, r1g1, mo, b0b, true);
    conv(b0b, r1w2, b1a, Xo, Yo, Zo, Xo, Yo, Zo, Cr, Cr, 3, -1, 1, 1, 1);
    const h16* sk;
    if (r1wd) {
      conv(p2, r1wd, b1b, Xo, Yo, Zo, Xo, Yo, Zo, Cin, Cr, 1, 0, 1, 1, 1);
      bn(b1b, V, Cr, r1bd, r1gd, mo, b1b, false);
      sk = b1b;
    } else sk = p2;
    bnAdd(b1a, sk, V, Cr, r1b2, r1g2, mo, p0);          // R1 in p0
    // res2: Cr -> Cr, identity skip
    conv(p0, r2w1, b0a, Xo, Yo, Zo, Xo, Yo, Zo, Cr, Cr, 3, -1, 1, 1, 1);
    bn(b0a, V, Cr, r2b1, r2g1, mo, b0b, true);
    conv(b0b, r2w2, b1a, Xo, Yo, Zo, Xo, Yo, Zo, Cr, Cr, 3, -1, 1, 1, 1);
    bnAdd(b1a, p0, V, Cr, r2b2, r2g2, mo, p1);          // stage out in p1
  };

  stage(big1, mask0, mask1, 256, 256, 32, 128, 128, 32, 2, 2, 1, 32, 64,
        0, 1, w_d1, 3, 4, 5, 6, 7, 8, w_r11, w_r12, w_r1d,
        12, 13, 14, 15, w_r21, w_r22);
  stage(p1, mask1, mask2, 128, 128, 32, 64, 64, 16, 2, 2, 2, 64, 128,
        18, 19, w_d2, 21, 22, 23, 24, 25, 26, w2_r11, w2_r12, w2_r1d,
        30, 31, 32, 33, w2_r21, w2_r22);
  stage(p1, mask2, mask3, 64, 64, 16, 32, 32, 8, 2, 2, 2, 128, 128,
        36, 37, w_d3, 39, 40, 41, 42, -1, -1, w3_r11, w3_r12, nullptr,
        45, 46, 47, 48, w3_r21, w3_r22);

  // ---- write [1,128,32,32,8] f32 channel-major
  k_final<<<(V3 * 128 + 255) / 256, 256, 0, stream>>>(p1, (float*)d_out,
                                                      ilog2(V3), V3 * 128, 7);
}
